// CausalVAE2_11733850652840
// MI455X (gfx1250) — matrix operand order repaired and checked by reading, not compiled, not run on hardware
//
#include <hip/hip_runtime.h>
#include <hip/hip_bf16.h>

#define DD   64
#define HID  256
#define BATCH 262144
#define B64  (BATCH * 64)

typedef __attribute__((ext_vector_type(16))) __bf16 v16bf;
typedef __attribute__((ext_vector_type(8)))  __bf16 v8bf;
typedef __attribute__((ext_vector_type(8)))  float  v8f;

__device__ __forceinline__ unsigned short bf16_bits(float f) {
    return __builtin_bit_cast(unsigned short, (__bf16)f);
}
__device__ __forceinline__ v16bf frag_from_2xb128(const uint4* p) {
    v8bf lo = __builtin_bit_cast(v8bf, p[0]);
    v8bf hi = __builtin_bit_cast(v8bf, p[1]);
    return __builtin_shufflevector(lo, hi, 0, 1, 2, 3, 4, 5, 6, 7,
                                   8, 9, 10, 11, 12, 13, 14, 15);
}
#define WMMA_BF16(A, B, C) \
    __builtin_amdgcn_wmma_f32_16x16x32_bf16(false, (A), false, (B), (short)0, (C), false, false)

// ---------------------------------------------------------------------------
// Prep 1: masked dec1_w -> bf16 in WMMA B-fragment lane order.
// frag index o = ((i*16 + nb)*32 + lane)*32 + kb*16 + e
// ---------------------------------------------------------------------------
__global__ __launch_bounds__(256) void prep_dec(const float* __restrict__ dec1_w,
                                                const unsigned char* __restrict__ Bmask,
                                                unsigned short* __restrict__ wfrag) {
    int o = blockIdx.x * 256 + threadIdx.x;          // 0 .. 64*16*32*32-1
    int e2   = o & 31;
    int kb   = e2 >> 4;
    int e    = e2 & 15;
    int lane = (o >> 5) & 31;
    int half = lane >> 4;
    int ln   = lane & 15;
    int nb   = (o >> 10) & 15;
    int i    = o >> 14;
    int k = kb * 32 + (e & 7) + 8 * half + 16 * (e >> 3); // B-frag K mapping
    int n = nb * 16 + ln;                            // B-frag N mapping
    float w = dec1_w[(i * 64 + k) * HID + n];
    bool  m = Bmask[k * 64 + i] != 0;                // mask_in[i][k] = B_mask[k][i]
    wfrag[o] = bf16_bits(m ? w : 0.0f);
}

// ---------------------------------------------------------------------------
// Prep 2: enc1_w / enc3_w -> bf16 B-fragment tables (32 KB + 64 KB).
// enc1 idx = ((nb*2+kb)*32+lane)*16+e ; enc3 idx = ((kb*8+j)*32+lane)*16+e
// ---------------------------------------------------------------------------
__global__ __launch_bounds__(256) void prep_enc(const float* __restrict__ enc1_w,
                                                const float* __restrict__ enc3_w,
                                                unsigned short* __restrict__ enc1frag,
                                                unsigned short* __restrict__ enc3frag) {
    int t = blockIdx.x * 256 + threadIdx.x;          // 0 .. 49151
    if (t < 16384) {
        int e = t & 15, lane = (t >> 4) & 31, kb = (t >> 9) & 1, nb = t >> 10;
        int half = lane >> 4, ln = lane & 15;
        int k = kb * 32 + (e & 7) + 8 * half + 16 * (e >> 3);
        enc1frag[t] = bf16_bits(enc1_w[k * HID + nb * 16 + ln]);
    } else {
        int u = t - 16384;
        int e = u & 15, lane = (u >> 4) & 31, j = (u >> 9) & 7, kb = u >> 12;
        int half = lane >> 4, ln = lane & 15;
        int k = kb * 32 + (e & 7) + 8 * half + 16 * (e >> 3);
        enc3frag[u] = bf16_bits(enc3_w[k * 128 + j * 16 + ln]);
    }
}

// ---------------------------------------------------------------------------
// Encoder: one wave per 16-sample tile, 4 waves/block.
// ---------------------------------------------------------------------------
__global__ __launch_bounds__(128) void enc_kernel(const float* __restrict__ x,
                                                  const float* __restrict__ eps,
                                                  const unsigned short* __restrict__ enc1frag,
                                                  const float* __restrict__ enc1_b,
                                                  const unsigned short* __restrict__ enc3frag,
                                                  const float* __restrict__ enc3_b,
                                                  float* __restrict__ out,
                                                  unsigned short* __restrict__ eps_frag) {
    __shared__ __align__(16) unsigned short hbuf[4][16 * 264];  // 16x256 bf16, padded
    __shared__ __align__(16) unsigned short ebuf[4][16 * 72];   // 16x64  bf16, padded
    const int wid  = threadIdx.x >> 5;
    const int lane = threadIdx.x & 31;
    const int half = lane >> 4;
    const int ln   = lane & 15;
    const long tile = (long)blockIdx.x * 4 + wid;
    const long r0   = tile * 16;

    // ---- A-fragments of x: per-lane K-runs are contiguous 8-float spans
    const float4* xr = (const float4*)(x + (r0 + ln) * 64);
    float4 p0 = xr[2 * half], p1 = xr[2 * half + 1];
    float4 p2 = xr[4 + 2 * half], p3 = xr[5 + 2 * half];
    float4 p4 = xr[8 + 2 * half], p5 = xr[9 + 2 * half];
    float4 p6 = xr[12 + 2 * half], p7 = xr[13 + 2 * half];
    v16bf a0, a1;
    {
        const float* f0 = (const float*)&p0;   // p0..p3 -> a0, p4..p7 -> a1
        const float* f1 = (const float*)&p4;
#pragma unroll
        for (int e = 0; e < 16; ++e) { a0[e] = (__bf16)f0[e]; a1[e] = (__bf16)f1[e]; }
    }

    // ---- layer 1: h = relu(x @ enc1_w + b1)
    unsigned short* h = hbuf[wid];
    for (int nb = 0; nb < 16; ++nb) {
        int n = nb * 16 + ln;
        v16bf b0 = frag_from_2xb128((const uint4*)(enc1frag + (nb * 2 + 0) * 512 + lane * 16));
        v16bf b1 = frag_from_2xb128((const uint4*)(enc1frag + (nb * 2 + 1) * 512 + lane * 16));
        v8f acc = {};
        acc = WMMA_BF16(a0, b0, acc);
        acc = WMMA_BF16(a1, b1, acc);
        float bias = enc1_b[n];
#pragma unroll
        for (int v = 0; v < 8; ++v) {
            float hv = fmaxf(acc[v] + bias, 0.0f);
            h[(v + 8 * half) * 264 + n] = bf16_bits(hv);
        }
    }
    __syncthreads();

    // ---- layer 2: z = h @ enc3_w
    v8f zacc[8];
#pragma unroll
    for (int j = 0; j < 8; ++j) zacc[j] = (v8f){};
    for (int kb = 0; kb < 8; ++kb) {
        // A-frag from LDS: two contiguous 16B runs per lane
        const unsigned short* hr = h + ln * 264 + kb * 32 + 8 * half;
        v8bf lo = __builtin_bit_cast(v8bf, *(const uint4*)hr);
        v8bf hi = __builtin_bit_cast(v8bf, *(const uint4*)(hr + 16));
        v16bf ha = __builtin_shufflevector(lo, hi, 0, 1, 2, 3, 4, 5, 6, 7,
                                           8, 9, 10, 11, 12, 13, 14, 15);
#pragma unroll
        for (int j = 0; j < 8; ++j) {
            v16bf wb = frag_from_2xb128((const uint4*)(enc3frag + (kb * 8 + j) * 512 + lane * 16));
            zacc[j] = WMMA_BF16(ha, wb, zacc[j]);
        }
    }

    // ---- mu / log_var / eps_y ; zero y_recon[:,64:]
    unsigned short* eb = ebuf[wid];
#pragma unroll
    for (int j = 0; j < 4; ++j) {
        int n = j * 16 + ln;                         // d in 0..63
        float bmu = enc3_b[n];
        float blv = enc3_b[64 + n];
#pragma unroll
        for (int v = 0; v < 8; ++v) {
            int M = v + 8 * half;
            long s = r0 + M;
            float mu = zacc[j][v] + bmu;
            float lv = zacc[j + 4][v] + blv;
            float ey = mu + expf(0.5f * lv) * eps[s * 64 + n];
            out[(long)B64 + s * 64 + n]           = mu;
            out[(long)2 * B64 + s * 64 + n]       = lv;
            out[(long)3 * B64 + s * 128 + 64 + n] = 0.0f;
            eb[M * 72 + n] = bf16_bits(ey);
        }
    }
    __syncthreads();

    // ---- emit eps_y A-fragments: 4 x ds_load_b128 -> 4 x global b128 store
    const unsigned short* er = eb + ln * 72 + 8 * half;
    uint4* dst = (uint4*)(eps_frag + (tile * 32 + lane) * 32);
    dst[0] = *(const uint4*)(er);
    dst[1] = *(const uint4*)(er + 16);
    dst[2] = *(const uint4*)(er + 32);
    dst[3] = *(const uint4*)(er + 48);
}

// ---------------------------------------------------------------------------
// Decoder (hot): 2 batch tiles per wave; all 64 heads reuse the A-fragments.
// Per (i, n-block): 4 WMMAs share one B-frag load pair + one bias/w3 load.
// ---------------------------------------------------------------------------
__global__ __launch_bounds__(256) void dec_kernel(const unsigned short* __restrict__ eps_frag,
                                                  const unsigned short* __restrict__ wfrag,
                                                  const float* __restrict__ dec1_b,
                                                  const float* __restrict__ dec3_w,
                                                  const float* __restrict__ dec3_b,
                                                  float* __restrict__ out) {
    const int wid  = threadIdx.x >> 5;
    const int lane = threadIdx.x & 31;
    const int half = lane >> 4;
    const int ln   = lane & 15;
    const long tile0 = ((long)blockIdx.x * 8 + wid) * 2;   // two consecutive tiles

    v16bf a[2][2];
#pragma unroll
    for (int t = 0; t < 2; ++t) {
        const uint4* asrc = (const uint4*)(eps_frag + ((tile0 + t) * 32 + lane) * 32);
        a[t][0] = frag_from_2xb128(asrc);
        a[t][1] = frag_from_2xb128(asrc + 2);
    }

    for (int i = 0; i < 64; ++i) {
        float partial[2][8];
#pragma unroll
        for (int t = 0; t < 2; ++t)
#pragma unroll
            for (int v = 0; v < 8; ++v) partial[t][v] = 0.0f;

        for (int nb = 0; nb < 16; ++nb) {
            const uint4* bsrc = (const uint4*)(wfrag + (((long)(i * 16 + nb) * 32 + lane) * 32));
            v16bf b0 = frag_from_2xb128(bsrc);
            v16bf b1 = frag_from_2xb128(bsrc + 2);

            v8f acc0 = {}, acc1 = {};
            acc0 = WMMA_BF16(a[0][0], b0, acc0);
            acc1 = WMMA_BF16(a[1][0], b0, acc1);
            acc0 = WMMA_BF16(a[0][1], b1, acc0);
            acc1 = WMMA_BF16(a[1][1], b1, acc1);

            int n = nb * 16 + ln;
            float bias = dec1_b[i * HID + n];
            float w3   = dec3_w[i * HID + n];
#pragma unroll
            for (int v = 0; v < 8; ++v) {
                partial[0][v] += fmaxf(acc0[v] + bias, 0.0f) * w3;
                partial[1][v] += fmaxf(acc1[v] + bias, 0.0f) * w3;
            }
        }
        // reduce the 256-wide dot across the 16 lanes of each half
#pragma unroll
        for (int m = 8; m >= 1; m >>= 1)
#pragma unroll
            for (int t = 0; t < 2; ++t)
#pragma unroll
                for (int v = 0; v < 8; ++v)
                    partial[t][v] += __shfl_xor(partial[t][v], m, 32);

        float b3 = dec3_b[i];
#pragma unroll
        for (int t = 0; t < 2; ++t) {
            float val = partial[t][0];
#pragma unroll
            for (int v = 1; v < 8; ++v) val = (ln == v) ? partial[t][v] : val;
            if (ln < 8) {
                long s = (tile0 + t) * 16 + ln + 8 * half;
                float o = val + b3;
                out[s * 64 + i]                  = o;   // x_out
                out[(long)3 * B64 + s * 128 + i] = o;   // y_recon[:, :64]
            }
        }
    }
}

extern "C" void kernel_launch(void* const* d_in, const int* in_sizes, int n_in,
                              void* d_out, int out_size, void* d_ws, size_t ws_size,
                              hipStream_t stream) {
    const float* x       = (const float*)d_in[0];
    const float* eps     = (const float*)d_in[1];
    const float* enc1_w  = (const float*)d_in[2];
    const float* enc1_b  = (const float*)d_in[3];
    const float* enc3_w  = (const float*)d_in[4];
    const float* enc3_b  = (const float*)d_in[5];
    const float* dec1_w  = (const float*)d_in[6];
    const float* dec1_b  = (const float*)d_in[7];
    const float* dec3_w  = (const float*)d_in[8];
    const float* dec3_b  = (const float*)d_in[9];
    const unsigned char* Bmask = (const unsigned char*)d_in[10];
    float* out = (float*)d_out;

    char* ws = (char*)d_ws;
    unsigned short* wfrag    = (unsigned short*)(ws);                    // 2 MB
    unsigned short* enc1frag = (unsigned short*)(ws + 2097152);          // 32 KB
    unsigned short* enc3frag = (unsigned short*)(ws + 2097152 + 32768);  // 64 KB
    unsigned short* eps_frag = (unsigned short*)(ws + 2097152 + 98304);  // 32 MB

    prep_dec<<<(64 * 16 * 32 * 32) / 256, 256, 0, stream>>>(dec1_w, Bmask, wfrag);
    prep_enc<<<192, 256, 0, stream>>>(enc1_w, enc3_w, enc1frag, enc3frag);
    enc_kernel<<<BATCH / 64, 128, 0, stream>>>(x, eps, enc1frag, enc1_b, enc3frag, enc3_b,
                                               out, eps_frag);
    dec_kernel<<<BATCH / 256, 256, 0, stream>>>(eps_frag, wfrag, dec1_b, dec3_w, dec3_b, out);
}
